// Autograd4bitQuantLinear_23587960390015
// MI455X (gfx1250) — matrix operand order repaired and checked by reading, not compiled, not run on hardware
//
#include <hip/hip_runtime.h>
#include <hip/hip_bf16.h>

typedef __attribute__((ext_vector_type(16))) _Float16 v16h;
typedef __attribute__((ext_vector_type(8)))  _Float16 v8h;
typedef __attribute__((ext_vector_type(8)))  float    v8f;

#define K_DIM 4096
#define N_DIM 11008
#define BM 128
#define BN 256
#define BK 32
#define LDP 40   // padded LDS row stride in f16 elements (80B = 20 banks, conflict-free)

// ---------------------------------------------------------------------------
// rowsum[m] = sum_k f32( f16( x[m][k] ) )  -- matches the precision the WMMA
// path actually multiplies, so the factored zero-point term is consistent.
// ---------------------------------------------------------------------------
__global__ __launch_bounds__(256)
void rowsum_kernel(const float* __restrict__ x, float* __restrict__ rs) {
    const int m = blockIdx.x;
    const int t = threadIdx.x;
    const float* row = x + (size_t)m * K_DIM;
    float s = 0.f;
    #pragma unroll 4
    for (int k = t; k < K_DIM; k += 256) {
        s += (float)(_Float16)row[k];
    }
    __shared__ float red[256];
    red[t] = s;
    __syncthreads();
    for (int off = 128; off > 0; off >>= 1) {
        if (t < off) red[t] += red[t + off];
        __syncthreads();
    }
    if (t == 0) rs[m] = red[0];
}

// ---------------------------------------------------------------------------
// Main GEMM: out[m][n] = s[n]*(x_f16 @ Q)[m][n] - z[n]*rowsum[m] + bias[n]
// Q = raw 4-bit nibbles (exact in f16). 128x256 tile / block, BK=32 steps,
// double-buffered LDS; 8 waves, each wave 64x64 = 4x4 wmma_f32_16x16x32_f16.
// ---------------------------------------------------------------------------
__global__ __launch_bounds__(256)
void gemm4bit_wmma(const float* __restrict__ x,
                   const int*   __restrict__ qweight,
                   const float* __restrict__ scales,
                   const float* __restrict__ zeros,
                   const float* __restrict__ bias,
                   const float* __restrict__ rowsum,
                   float*       __restrict__ out)
{
    __shared__ _Float16 ldsA[2][BM * LDP];   // A tile, row-major [m][k], f16   (20 KB)
    __shared__ _Float16 ldsB[2][BN * LDP];   // W tile, col-major  [n][k], f16  (40 KB)

    const int tid  = threadIdx.x;
    const int n0   = blockIdx.x * BN;
    const int m0   = blockIdx.y * BM;

    const int wave  = tid >> 5;
    const int lane  = tid & 31;
    const int lrow  = lane & 15;         // 0..15
    const int lhi   = lane >> 4;         // 0/1 : half-wave
    const int waveM = (wave & 1) * 64;   // 2 waves along M
    const int waveN = (wave >> 1) * 64;  // 4 waves along N

    // ---- global-load staging registers (double buffering) ----
    float4 aReg[4];   // 16 f32 of x per thread per K-step
    int    wReg[4];   // 4 packed int32 (32 nibbles) per thread per K-step

    const int ch0 = tid * 2;  // A chunks: 512 chunks of 8 f32 (128 rows x 4 chunks)

    auto gloadA = [&](int kt) {
        const float* base = x + (size_t)m0 * K_DIM + kt * BK;
        #pragma unroll
        for (int i = 0; i < 2; ++i) {
            const int ch = ch0 + i;
            const int r  = ch >> 2;
            const int cc = (ch & 3) * 8;
            const float4* p = (const float4*)(base + (size_t)r * K_DIM + cc);
            aReg[i * 2 + 0] = p[0];
            aReg[i * 2 + 1] = p[1];
        }
    };
    auto gloadW = [&](int kt) {
        // 1024 ints per step: packed row pr = i (0..3), column n = tid (0..255)
        const int kr0 = kt * (BK / 8);
        #pragma unroll
        for (int i = 0; i < 4; ++i) {
            wReg[i] = qweight[(size_t)(kr0 + i) * N_DIM + n0 + tid];
        }
    };
    auto stage_store = [&](int st) {
        // x f32 -> f16, b128 stores
        #pragma unroll
        for (int i = 0; i < 2; ++i) {
            const int ch = ch0 + i;
            const int r  = ch >> 2;
            const int cc = (ch & 3) * 8;
            const float* f = (const float*)&aReg[i * 2];
            v8h v;
            #pragma unroll
            for (int j = 0; j < 8; ++j) v[j] = (_Float16)f[j];
            *(v8h*)&ldsA[st][r * LDP + cc] = v;
        }
        // nibble -> f16 (exact), stored transposed: ldsB[n][k]
        #pragma unroll
        for (int i = 0; i < 4; ++i) {
            const unsigned g = (unsigned)wReg[i];
            v8h v;
            #pragma unroll
            for (int j = 0; j < 8; ++j)
                v[j] = (_Float16)(int)((g >> (4 * j)) & 0xF);
            *(v8h*)&ldsB[st][tid * LDP + i * 8] = v;
        }
    };

    v8f acc[4][4];
    #pragma unroll
    for (int mi = 0; mi < 4; ++mi)
        #pragma unroll
        for (int ni = 0; ni < 4; ++ni)
            acc[mi][ni] = (v8f)(0.f);

    gloadA(0); gloadW(0);
    stage_store(0);
    __syncthreads();

    const int NKT = K_DIM / BK;   // 128
    for (int kt = 0; kt < NKT; ++kt) {
        const int cur = kt & 1;
        if (kt + 1 < NKT) { gloadA(kt + 1); gloadW(kt + 1); }

        // ---- A fragments: 16x32 f16 layout ----
        // lanes 0-15 (row M=lrow):  K = [0..7] and [16..23]
        // lanes 16-31 (row M=lrow): K = [8..15] and [24..31]
        v16h afrag[4], bfrag[4];
        #pragma unroll
        for (int mi = 0; mi < 4; ++mi) {
            const int row = waveM + mi * 16 + lrow;
            const _Float16* pa = &ldsA[cur][row * LDP + lhi * 8];
            v8h lo = *(const v8h*)pa;
            v8h hi = *(const v8h*)(pa + 16);
            #pragma unroll
            for (int j = 0; j < 8; ++j) { afrag[mi][j] = lo[j]; afrag[mi][j + 8] = hi[j]; }
        }
        // ---- B fragments: 32x16 f16, lane = column n ----
        // lanes 0-15: K=[0..7] and [16..23] of column lrow ; lanes 16-31: K=[8..15] and [24..31]
        #pragma unroll
        for (int ni = 0; ni < 4; ++ni) {
            const int col = waveN + ni * 16 + lrow;
            const _Float16* pb = &ldsB[cur][col * LDP + lhi * 8];
            v8h lo = *(const v8h*)pb;
            v8h hi = *(const v8h*)(pb + 16);
            #pragma unroll
            for (int j = 0; j < 8; ++j) { bfrag[ni][j] = lo[j]; bfrag[ni][j + 8] = hi[j]; }
        }

        #pragma unroll
        for (int mi = 0; mi < 4; ++mi)
            #pragma unroll
            for (int ni = 0; ni < 4; ++ni)
                acc[mi][ni] = __builtin_amdgcn_wmma_f32_16x16x32_f16(
                    false, afrag[mi], false, bfrag[ni],
                    (short)0, acc[mi][ni], false, false);

        if (kt + 1 < NKT) stage_store(1 - cur);
        __syncthreads();
    }

    // ---- epilogue: C/D layout: VGPR g -> M = g + 8*lhi, N = lrow ----
    #pragma unroll
    for (int mi = 0; mi < 4; ++mi) {
        const int mBase = m0 + waveM + mi * 16 + lhi * 8;
        #pragma unroll
        for (int ni = 0; ni < 4; ++ni) {
            const int n = n0 + waveN + ni * 16 + lrow;
            const float s  = scales[n];
            const float z  = zeros[n];
            const float bv = bias[n];
            #pragma unroll
            for (int g = 0; g < 8; ++g) {
                const int m = mBase + g;
                const float r = rowsum[m];
                const float v = s * acc[mi][ni][g] - z * r + bv;
                // stream the 360MB output past L2 so x/qweight stay resident
                __builtin_nontemporal_store(v, &out[(size_t)m * N_DIM + n]);
            }
        }
    }
}

extern "C" void kernel_launch(void* const* d_in, const int* in_sizes, int n_in,
                              void* d_out, int out_size, void* d_ws, size_t ws_size,
                              hipStream_t stream) {
    const float* x      = (const float*)d_in[0];  // (4,2048,4096) f32
    const int*   qw     = (const int*)  d_in[1];  // (512,11008)   i32
    const float* scales = (const float*)d_in[2];  // (11008,1)     f32
    const float* zeros  = (const float*)d_in[3];  // (11008,1)     f32
    const float* bias   = (const float*)d_in[4];  // (11008,)      f32
    float*       out    = (float*)d_out;          // (4,2048,11008) f32
    float*       rowsum = (float*)d_ws;           // M floats scratch

    const int M = in_sizes[0] / K_DIM;            // 8192

    rowsum_kernel<<<dim3(M), dim3(256), 0, stream>>>(x, rowsum);

    dim3 grid(N_DIM / BN, M / BM);                // 43 x 64
    gemm4bit_wmma<<<grid, dim3(256), 0, stream>>>(x, qw, scales, zeros, bias,
                                                  rowsum, out);
}
